// GAT_43361989821044
// MI455X (gfx1250) — matrix operand order repaired and checked by reading, not compiled, not run on hardware
//
#include <hip/hip_runtime.h>
#include <hip/hip_fp16.h>

typedef __attribute__((ext_vector_type(16))) _Float16 v16h;
typedef __attribute__((ext_vector_type(8)))  _Float16 v8h;
typedef __attribute__((ext_vector_type(8)))  float    v8f;
typedef __attribute__((ext_vector_type(8)))  int      v8i;

#define WMMA_F16(a, b, c) \
  __builtin_amdgcn_wmma_f32_16x16x32_f16(false, (a), false, (b), (short)0, (c), false, false)

namespace {
constexpr int B = 4, N = 2048, F = 256, D = 64, H = 4, C = 32, K1 = 256;
constexpr float LALPHA = 0.2f;
constexpr float NEGV   = -9e15f;
constexpr int NSPLIT = 8;            // split-K over source dim in output attention
constexpr int JSP = N / NSPLIT;      // 256 sources per split
}

// ---- CDNA5 async global->LDS helpers (ASYNCcnt path, CDNA5 ISA) ----
__device__ __forceinline__ void async_load_b128(const void* gptr, void* lptr) {
  unsigned lds = (unsigned)(uintptr_t)lptr;               // low 32 bits = LDS addr
  asm volatile("global_load_async_to_lds_b128 %0, %1, off"
               :: "v"(lds), "v"((unsigned long long)(uintptr_t)gptr) : "memory");
}
__device__ __forceinline__ void async_load_b32(const void* gptr, void* lptr) {
  unsigned lds = (unsigned)(uintptr_t)lptr;
  asm volatile("global_load_async_to_lds_b32 %0, %1, off"
               :: "v"(lds), "v"((unsigned long long)(uintptr_t)gptr) : "memory");
}
__device__ __forceinline__ void wait_async0() {
#if __has_builtin(__builtin_amdgcn_s_wait_asynccnt)
  __builtin_amdgcn_s_wait_asynccnt(0);
#else
  asm volatile("s_wait_asynccnt 0x0" ::: "memory");
#endif
}

// ---------------------------------------------------------------------------
// Kernel 0: weight prep. WT[h][d][f] = f16(W[h][f][d]); WoT[c][k] = f16(Wo[k][c])
// Puts every WMMA B operand in K-contiguous f16 so tiles load as v16h.
// ---------------------------------------------------------------------------
__global__ void prep_weights_kernel(const float* __restrict__ W,
                                    const float* __restrict__ Wo,
                                    _Float16* __restrict__ WT,
                                    _Float16* __restrict__ WoT) {
  int i = blockIdx.x * blockDim.x + threadIdx.x;
  if (i < H * D * F) {
    int f = i % F, d = (i / F) % D, h = i / (F * D);
    WT[i] = (_Float16)W[((size_t)h * F + f) * D + d];
  }
  if (i < C * H * D) {
    int k = i % (H * D), c = i / (H * D);
    WoT[i] = (_Float16)Wo[(size_t)k * C + c];
  }
}

// ---------------------------------------------------------------------------
// Kernel 1: Wh[b,h] = x[b] @ W[h] -> stored TRANSPOSED whT[b,h][d][n] f16.
// A: x tile via 2x v8f vector loads (+cvt). B: WT tile via one v16h load.
// ---------------------------------------------------------------------------
__global__ __launch_bounds__(32) void wh_heads_kernel(
    const float* __restrict__ x, const _Float16* __restrict__ WT,
    _Float16* __restrict__ whT) {
  int blk   = blockIdx.x;
  int itile = blk % (N / 16);
  int bh    = blk / (N / 16);
  int h = bh % H, b = bh / H;
  int i0 = itile * 16;
  int lane = threadIdx.x & 31;
  int M = lane & 15, hs = lane >> 4;
  int kAoff = hs * 16, kBoff = hs * 16;

  v8f acc[4] = {};
  size_t xbase = ((size_t)b * N + (i0 + M)) * F;

  for (int kk = 0; kk < F / 32; ++kk) {
    int k0 = kk * 32;
    v8f r0 = *(const v8f*)(x + xbase + k0 + kAoff);        // K = kAoff+0..7
    v8f r1 = *(const v8f*)(x + xbase + k0 + kAoff + 8);    // K = kAoff+8..15
    v16h a;
#pragma unroll
    for (int q = 0; q < 8; ++q) { a[q] = (_Float16)r0[q]; a[8 + q] = (_Float16)r1[q]; }
#pragma unroll
    for (int t = 0; t < 4; ++t) {
      int col = t * 16 + M;
      v16h bv = *(const v16h*)(WT + ((size_t)h * D + col) * F + k0 + kBoff);
      acc[t] = WMMA_F16(a, bv, acc[t]);
    }
  }
#pragma unroll
  for (int t = 0; t < 4; ++t)
#pragma unroll
    for (int v = 0; v < 8; ++v) {
      int row = v + 8 * hs, col = t * 16 + M;
      whT[(((size_t)(b * H + h)) * D + col) * N + i0 + row] = (_Float16)acc[t][v];
    }
}

// ---------------------------------------------------------------------------
// Kernel 2: f1/f2 rank-1 projections (coalesced reads of transposed whT)
// ---------------------------------------------------------------------------
__global__ void f12_heads_kernel(const _Float16* __restrict__ whT,
                                 const float* __restrict__ a1,
                                 const float* __restrict__ a2,
                                 float* __restrict__ f1, float* __restrict__ f2) {
  int r = blockIdx.x * blockDim.x + threadIdx.x;   // over B*H*N
  if (r >= B * H * N) return;
  int n = r % N, bh = r / N, h = bh % H;
  float s1 = 0.f, s2 = 0.f;
  for (int d = 0; d < D; ++d) {
    float v = (float)whT[((size_t)bh * D + d) * N + n];
    s1 += v * a1[h * D + d];
    s2 += v * a2[h * D + d];
  }
  f1[r] = s1; f2[r] = s2;
}

// ---------------------------------------------------------------------------
// Kernel 3: heads attention, flash-style. Block = 4 waves = 4 heads sharing
// one (b, 16-row) query tile. adj tile staged ONCE per block through the
// CDNA5 async global->LDS path and reused by all heads.
// ---------------------------------------------------------------------------
__global__ __launch_bounds__(128) void attn_heads_kernel(
    const int* __restrict__ adj, const _Float16* __restrict__ whT,
    const float* __restrict__ f1g, const float* __restrict__ f2g,
    _Float16* __restrict__ hcat) {
  __shared__ __align__(32) int   adjt[16 * 32];
  __shared__ __align__(32) float f2s[H][32];

  int b  = blockIdx.x / (N / 16);
  int i0 = (blockIdx.x % (N / 16)) * 16;
  int h    = threadIdx.x >> 5;     // wave id == head
  int lane = threadIdx.x & 31;
  int M = lane & 15, hs = lane >> 4;
  size_t bh = (size_t)b * H + h;

  float f1v = f1g[bh * N + i0 + M];
  float m = NEGV, l = 0.f;
  v8f acc[4] = {};
  int kAoff = hs * 16, kBoff = hs * 16;

  // per-thread 16B slice of the 2KB adj tile
  int arr = threadIdx.x >> 3, acc4 = (threadIdx.x & 7) * 4;
  size_t adjbase = ((size_t)b * N + i0 + arr) * N + acc4;

  for (int j0 = 0; j0 < N; j0 += 32) {
    // Async-stage adj tile (shared by all 4 heads) + per-head f2 chunk.
    async_load_b128(adj + adjbase + j0, &adjt[threadIdx.x * 4]);
    async_load_b32(f2g + bh * N + j0 + lane, &f2s[h][lane]);
    wait_async0();
    __syncthreads();

    // Scores e = leakyrelu(f1_i + f2_j), masked; WMMA A tiling order.
    v8i ma = *(const v8i*)&adjt[M * 32 + kAoff];
    v8i mb = *(const v8i*)&adjt[M * 32 + kAoff + 8];
    v8f fa = *(const v8f*)&f2s[h][kAoff];
    v8f fb = *(const v8f*)&f2s[h][kAoff + 8];
    float e[16];
    float cm = NEGV;
#pragma unroll
    for (int q = 0; q < 8; ++q) {
      float s0 = f1v + fa[q];
      float l0 = s0 > 0.f ? s0 : LALPHA * s0;
      float e0 = ma[q] > 0 ? l0 : NEGV;
      float s1 = f1v + fb[q];
      float l1 = s1 > 0.f ? s1 : LALPHA * s1;
      float e1 = mb[q] > 0 ? l1 : NEGV;
      e[q] = e0; e[8 + q] = e1;
      cm = fmaxf(cm, fmaxf(e0, e1));
    }
    cm = fmaxf(cm, __shfl_xor(cm, 16));      // row split across lane & lane+16
    float mn    = fmaxf(m, cm);
    float alpha = __expf(m - mn);
    float rs = 0.f;
    v16h a;
#pragma unroll
    for (int q = 0; q < 16; ++q) {
      float p = __expf(e[q] - mn);
      rs += p;
      a[q] = (_Float16)p;
    }
    rs += __shfl_xor(rs, 16);
    l = l * alpha + rs;
    m = mn;

    float arow[8];
#pragma unroll
    for (int v = 0; v < 8; ++v) arow[v] = __shfl(alpha, v + 8 * hs);
#pragma unroll
    for (int t = 0; t < 4; ++t)
#pragma unroll
      for (int v = 0; v < 8; ++v) acc[t][v] *= arow[v];

    // acc += P(16x32) @ Wh(32x64): B tile = one v16h load from transposed whT.
#pragma unroll
    for (int t = 0; t < 4; ++t) {
      int col = t * 16 + M;
      v16h bv = *(const v16h*)(whT + (bh * D + col) * N + j0 + kBoff);
      acc[t] = WMMA_F16(a, bv, acc[t]);
    }
    __syncthreads();  // protect adjt/f2s before next chunk's staging
  }

  float linv = 1.f / l;
  float lrow[8];
#pragma unroll
  for (int v = 0; v < 8; ++v) lrow[v] = __shfl(linv, v + 8 * hs);
#pragma unroll
  for (int t = 0; t < 4; ++t)
#pragma unroll
    for (int v = 0; v < 8; ++v) {
      float hv = acc[t][v] * lrow[v];
      hv = hv > 0.f ? hv : __expf(hv) - 1.f;     // elu
      int row = v + 8 * hs, col = t * 16 + M;
      hcat[((size_t)b * N + i0 + row) * (H * D) + h * D + col] = (_Float16)hv;
    }
}

// ---------------------------------------------------------------------------
// Kernel 4: Wh2 = hcat @ Wo -> stored TRANSPOSED wh2T[b][c][n] f16.
// A: hcat rows via 2x v8h loads. B: WoT tile via one v16h load.
// ---------------------------------------------------------------------------
__global__ __launch_bounds__(32) void wh_out_kernel(
    const _Float16* __restrict__ hcat, const _Float16* __restrict__ WoT,
    _Float16* __restrict__ wh2T) {
  int b  = blockIdx.x / (N / 16);
  int i0 = (blockIdx.x % (N / 16)) * 16;
  int lane = threadIdx.x & 31;
  int M = lane & 15, hs = lane >> 4;
  int kAoff = hs * 16, kBoff = hs * 16;

  v8f acc[2] = {};
  size_t abase = ((size_t)b * N + i0 + M) * (H * D);
  for (int kk = 0; kk < (H * D) / 32; ++kk) {
    int k0 = kk * 32;
    v8h r0 = *(const v8h*)(hcat + abase + k0 + kAoff);
    v8h r1 = *(const v8h*)(hcat + abase + k0 + kAoff + 8);
    v16h a;
#pragma unroll
    for (int q = 0; q < 8; ++q) { a[q] = r0[q]; a[8 + q] = r1[q]; }
#pragma unroll
    for (int t = 0; t < 2; ++t) {
      int col = t * 16 + M;
      v16h bv = *(const v16h*)(WoT + (size_t)col * (H * D) + k0 + kBoff);
      acc[t] = WMMA_F16(a, bv, acc[t]);
    }
  }
#pragma unroll
  for (int t = 0; t < 2; ++t)
#pragma unroll
    for (int v = 0; v < 8; ++v) {
      int row = v + 8 * hs, col = t * 16 + M;
      wh2T[((size_t)b * C + col) * N + i0 + row] = (_Float16)acc[t][v];
    }
}

// ---------------------------------------------------------------------------
// Kernel 5: f1o/f2o projections (coalesced reads of transposed wh2T)
// ---------------------------------------------------------------------------
__global__ void f12_out_kernel(const _Float16* __restrict__ wh2T,
                               const float* __restrict__ ao1,
                               const float* __restrict__ ao2,
                               float* __restrict__ f1o, float* __restrict__ f2o) {
  int r = blockIdx.x * blockDim.x + threadIdx.x;  // over B*N
  if (r >= B * N) return;
  int n = r % N, b = r / N;
  float s1 = 0.f, s2 = 0.f;
  for (int c = 0; c < C; ++c) {
    float v = (float)wh2T[((size_t)b * C + c) * N + n];
    s1 += v * ao1[c];
    s2 += v * ao2[c];
  }
  f1o[r] = s1; f2o[r] = s2;
}

// ---------------------------------------------------------------------------
// Kernel 6: output attention, only for gathered query rows, split-K over
// sources -> flash partials. Vectorized adj/f2 loads, v16h B tiles.
// ---------------------------------------------------------------------------
__global__ __launch_bounds__(32) void attn_out_kernel(
    const int* __restrict__ adj, const int* __restrict__ idcs,
    const _Float16* __restrict__ wh2T, const float* __restrict__ f1g,
    const float* __restrict__ f2g, float* __restrict__ pm,
    float* __restrict__ pl, float* __restrict__ pacc) {
  int s  = blockIdx.x % NSPLIT;
  int kt = (blockIdx.x / NSPLIT) % (K1 / 16);
  int b  = blockIdx.x / (NSPLIT * (K1 / 16));
  int lane = threadIdx.x & 31;
  int M = lane & 15, hs = lane >> 4;
  int kAoff = hs * 16, kBoff = hs * 16;

  int qi = idcs[b * K1 + kt * 16 + M];
  float f1v = f1g[(size_t)b * N + qi];
  size_t adjrow = ((size_t)b * N + qi) * N;
  const float* f2b = f2g + (size_t)b * N;

  float m = NEGV, l = 0.f;
  v8f acc[2] = {};
  int jbeg = s * JSP, jend = jbeg + JSP;

  for (int j0 = jbeg; j0 < jend; j0 += 32) {
    v8i ma = *(const v8i*)(adj + adjrow + j0 + kAoff);
    v8i mb = *(const v8i*)(adj + adjrow + j0 + kAoff + 8);
    v8f fa = *(const v8f*)(f2b + j0 + kAoff);
    v8f fb = *(const v8f*)(f2b + j0 + kAoff + 8);
    float e[16];
    float cm = NEGV;
#pragma unroll
    for (int q = 0; q < 8; ++q) {
      float s0 = f1v + fa[q];
      float l0 = s0 > 0.f ? s0 : LALPHA * s0;
      float e0 = ma[q] > 0 ? l0 : NEGV;
      float s1 = f1v + fb[q];
      float l1 = s1 > 0.f ? s1 : LALPHA * s1;
      float e1 = mb[q] > 0 ? l1 : NEGV;
      e[q] = e0; e[8 + q] = e1;
      cm = fmaxf(cm, fmaxf(e0, e1));
    }
    cm = fmaxf(cm, __shfl_xor(cm, 16));
    float mn    = fmaxf(m, cm);
    float alpha = __expf(m - mn);
    float rs = 0.f;
    v16h a;
#pragma unroll
    for (int q = 0; q < 16; ++q) {
      float p = __expf(e[q] - mn);
      rs += p;
      a[q] = (_Float16)p;
    }
    rs += __shfl_xor(rs, 16);
    l = l * alpha + rs;
    m = mn;

    float arow[8];
#pragma unroll
    for (int v = 0; v < 8; ++v) arow[v] = __shfl(alpha, v + 8 * hs);
#pragma unroll
    for (int t = 0; t < 2; ++t)
#pragma unroll
      for (int v = 0; v < 8; ++v) acc[t][v] *= arow[v];

#pragma unroll
    for (int t = 0; t < 2; ++t) {
      int col = t * 16 + M;
      v16h bv = *(const v16h*)(wh2T + ((size_t)b * C + col) * N + j0 + kBoff);
      acc[t] = WMMA_F16(a, bv, acc[t]);
    }
  }

  int rg = b * K1 + kt * 16;
  if (lane < 16) {
    pm[(size_t)(rg + M) * NSPLIT + s] = m;
    pl[(size_t)(rg + M) * NSPLIT + s] = l;
  }
#pragma unroll
  for (int t = 0; t < 2; ++t)
#pragma unroll
    for (int v = 0; v < 8; ++v) {
      int row = v + 8 * hs, col = t * 16 + M;
      pacc[((size_t)(rg + row) * NSPLIT + s) * C + col] = acc[t][v];
    }
}

// ---------------------------------------------------------------------------
// Kernel 7: merge flash partials, ELU, classifier (Linear->PReLU->Linear)
// ---------------------------------------------------------------------------
__global__ void merge_cls_kernel(const float* __restrict__ pm,
                                 const float* __restrict__ pl,
                                 const float* __restrict__ pacc,
                                 const float* __restrict__ W1,
                                 const float* __restrict__ b1,
                                 const float* __restrict__ pw,
                                 const float* __restrict__ W2,
                                 const float* __restrict__ b2,
                                 float* __restrict__ out) {
  int r = blockIdx.x * blockDim.x + threadIdx.x;   // over B*K1
  if (r >= B * K1) return;
  float ms = NEGV;
  for (int s = 0; s < NSPLIT; ++s) ms = fmaxf(ms, pm[(size_t)r * NSPLIT + s]);
  float Lsum = 0.f;
  float ef[C];
  for (int c = 0; c < C; ++c) ef[c] = 0.f;
  for (int s = 0; s < NSPLIT; ++s) {
    float w = __expf(pm[(size_t)r * NSPLIT + s] - ms);
    Lsum += pl[(size_t)r * NSPLIT + s] * w;
    const float* pa = pacc + ((size_t)r * NSPLIT + s) * C;
    for (int c = 0; c < C; ++c) ef[c] += pa[c] * w;
  }
  float Linv = 1.f / Lsum;
  for (int c = 0; c < C; ++c) {
    float v = ef[c] * Linv;
    ef[c] = v > 0.f ? v : __expf(v) - 1.f;         // elu on GAT output
  }
  float y[C];
  for (int c = 0; c < C; ++c) {
    float s1 = b1[c];
    for (int d = 0; d < C; ++d) s1 += ef[d] * W1[d * C + c];
    y[c] = s1 > 0.f ? s1 : pw[c] * s1;             // PReLU
  }
  for (int o = 0; o < 2; ++o) {
    float s2 = b2[o];
    for (int c = 0; c < C; ++c) s2 += y[c] * W2[c * 2 + o];
    out[(size_t)r * 2 + o] = s2;
  }
}

// ---------------------------------------------------------------------------
extern "C" void kernel_launch(void* const* d_in, const int* in_sizes, int n_in,
                              void* d_out, int out_size, void* d_ws, size_t ws_size,
                              hipStream_t stream) {
  (void)in_sizes; (void)n_in; (void)out_size; (void)ws_size;
  const float* x    = (const float*)d_in[0];
  const int*   adj  = (const int*)d_in[1];
  const int*   idcs = (const int*)d_in[2];
  const float* W    = (const float*)d_in[3];
  const float* a1   = (const float*)d_in[4];
  const float* a2   = (const float*)d_in[5];
  const float* Wo   = (const float*)d_in[6];
  const float* ao1  = (const float*)d_in[7];
  const float* ao2  = (const float*)d_in[8];
  const float* W1   = (const float*)d_in[9];
  const float* b1   = (const float*)d_in[10];
  const float* pw   = (const float*)d_in[11];
  const float* W2   = (const float*)d_in[12];
  const float* b2   = (const float*)d_in[13];
  float* out = (float*)d_out;

  char* ws = (char*)d_ws;
  size_t off = 0;
  auto alloc = [&](size_t bytes) -> char* {
    char* p = ws + off;
    off += (bytes + 255) & ~(size_t)255;
    return p;
  };
  _Float16* WT   = (_Float16*)alloc((size_t)H * D * F * 2);       // 128 KB
  _Float16* WoT  = (_Float16*)alloc((size_t)C * H * D * 2);       // 16 KB
  _Float16* whT  = (_Float16*)alloc((size_t)B * H * D * N * 2);   // 4 MB
  float*    f1   = (float*)alloc((size_t)B * H * N * 4);
  float*    f2   = (float*)alloc((size_t)B * H * N * 4);
  _Float16* hcat = (_Float16*)alloc((size_t)B * N * H * D * 2);   // 4 MB
  _Float16* wh2T = (_Float16*)alloc((size_t)B * C * N * 2);
  float*    f1o  = (float*)alloc((size_t)B * N * 4);
  float*    f2o  = (float*)alloc((size_t)B * N * 4);
  float*    pm   = (float*)alloc((size_t)B * K1 * NSPLIT * 4);
  float*    pl   = (float*)alloc((size_t)B * K1 * NSPLIT * 4);
  float*    pacc = (float*)alloc((size_t)B * K1 * NSPLIT * C * 4);

  prep_weights_kernel<<<dim3((H * D * F + 255) / 256), dim3(256), 0, stream>>>(W, Wo, WT, WoT);
  wh_heads_kernel<<<dim3(B * H * (N / 16)), dim3(32), 0, stream>>>(x, WT, whT);
  f12_heads_kernel<<<dim3((B * H * N + 127) / 128), dim3(128), 0, stream>>>(whT, a1, a2, f1, f2);
  attn_heads_kernel<<<dim3(B * (N / 16)), dim3(128), 0, stream>>>(adj, whT, f1, f2, hcat);
  wh_out_kernel<<<dim3(B * (N / 16)), dim3(32), 0, stream>>>(hcat, WoT, wh2T);
  f12_out_kernel<<<dim3((B * N + 127) / 128), dim3(128), 0, stream>>>(wh2T, ao1, ao2, f1o, f2o);
  attn_out_kernel<<<dim3(B * (K1 / 16) * NSPLIT), dim3(32), 0, stream>>>(
      adj, idcs, wh2T, f1o, f2o, pm, pl, pacc);
  merge_cls_kernel<<<dim3((B * K1 + 127) / 128), dim3(128), 0, stream>>>(
      pm, pl, pacc, W1, b1, pw, W2, b2, out);
}
